// EdgeTopKPooling_81088982548801
// MI455X (gfx1250) — matrix operand order repaired and checked by reading, not compiled, not run on hardware
//
#include <hip/hip_runtime.h>

// ---------------- problem constants (from reference) ----------------
#define B_GRAPHS 16
#define N_PER    25000
#define N_TOT    (B_GRAPHS * N_PER)   // 400000
#define C_DIM    256
#define K_KEEP   12500
#define BK       (B_GRAPHS * K_KEEP)  // 200000
#define SORT_P2  32768                // next pow2 >= N_PER

typedef __attribute__((ext_vector_type(16))) _Float16 v16h;
typedef __attribute__((ext_vector_type(8)))  float    v8f;
typedef __attribute__((ext_vector_type(4)))  float    f4;

// LDS byte offset of a __shared__ object: AS3->generic cast places the LDS
// offset in the low 32 bits of the flat address.
__device__ __forceinline__ unsigned lds_offset_of(const void* p) {
    return (unsigned)(unsigned long long)p;
}

// Async DMA: global B32 -> LDS (GV mode, no SADDR). Tracked by ASYNCcnt.
__device__ __forceinline__ void async_load_b32_to_lds(unsigned lds_off, const void* gaddr) {
    asm volatile("global_load_async_to_lds_b32 %0, %1, off"
                 :: "v"(lds_off), "v"(gaddr) : "memory");
}
__device__ __forceinline__ void wait_asynccnt0() {
    asm volatile("s_wait_asynccnt 0x0" ::: "memory");
}

// ---------------- 1) inv_norm = 1/||w||2 ----------------
__global__ void k_wnorm(const float* __restrict__ w, float* __restrict__ inv_norm) {
    __shared__ float red[256];
    int t = threadIdx.x;
    float v = w[t];
    red[t] = v * v;
    __syncthreads();
    for (int off = 128; off > 0; off >>= 1) {
        if (t < off) red[t] += red[t + off];
        __syncthreads();
    }
    if (t == 0) inv_norm[0] = rsqrtf(red[0]);
}

// ---------------- 2) score[i] = tanh(dot(ea[i,:], w) * inv_norm) via WMMA ----------------
// One wave = one 16-row tile. A: 16x32 f16 tile of edge_attr rows; B: 32x16 f16 with
// w broadcast into every column => D[m,n] == dot(row m, w-chunk) for every n.
// A layout (16-bit A 16x32): lane<16 holds K = {h<8 ? h : h+8}; lane>=16 adds +8.
// B layout (16-bit B 32x16): same K order as A: lane<16 holds K = {h<8 ? h : h+8}; lane>=16 adds +8.
// C/D f32 16x16 layout: VGPR v, lane 0 -> (M=v, N=0); lane 16 -> (M=8+v, N=0).
// Weight is DMA-staged into LDS once per block (ASYNCcnt path).
__global__ void k_score_wmma(const float* __restrict__ ea, const float* __restrict__ w,
                             const float* __restrict__ inv_norm, float* __restrict__ score) {
    __shared__ float w_lds[C_DIM];

    // one async B32 per thread: w[0..255] -> LDS, no VGPR round-trip
    async_load_b32_to_lds(lds_offset_of(&w_lds[threadIdx.x]), w + threadIdx.x);
    wait_asynccnt0();
    __syncthreads();

    const int lane = threadIdx.x & 31;
    const int wave = threadIdx.x >> 5;
    const int tile = blockIdx.x * 8 + wave;
    const int hi   = lane >> 4;                 // 0 or 1
    const float* rp = ea + (size_t)(tile * 16 + (lane & 15)) * C_DIM;

    v8f acc = {};
#pragma unroll 4
    for (int chunk = 0; chunk < 8; ++chunk) {
        const int k0 = chunk * 32;
        // --- A: two 8-wide runs of this lane's row, converted f32->f16 ---
        const int abase = k0 + hi * 8;
        f4 fa0 = *(const f4*)(rp + abase);
        f4 fa1 = *(const f4*)(rp + abase + 4);
        f4 fa2 = *(const f4*)(rp + abase + 16);
        f4 fa3 = *(const f4*)(rp + abase + 20);
        v16h a;
        a[0]=(_Float16)fa0.x;  a[1]=(_Float16)fa0.y;  a[2]=(_Float16)fa0.z;  a[3]=(_Float16)fa0.w;
        a[4]=(_Float16)fa1.x;  a[5]=(_Float16)fa1.y;  a[6]=(_Float16)fa1.z;  a[7]=(_Float16)fa1.w;
        a[8]=(_Float16)fa2.x;  a[9]=(_Float16)fa2.y;  a[10]=(_Float16)fa2.z; a[11]=(_Float16)fa2.w;
        a[12]=(_Float16)fa3.x; a[13]=(_Float16)fa3.y; a[14]=(_Float16)fa3.z; a[15]=(_Float16)fa3.w;
        // --- B: weight chunk (from LDS) broadcast across all 16 columns ---
        const int bbase = k0 + hi * 8;
        v16h b;
#pragma unroll
        for (int h = 0; h < 16; ++h) b[h] = (_Float16)w_lds[bbase + (h < 8 ? h : h + 8)];
        // (neg_a, A, neg_b, B, c_mod, C, reuse_a, reuse_b)
        acc = __builtin_amdgcn_wmma_f32_16x16x32_f16(false, a, false, b,
                                                     (short)0, acc, false, false);
    }
    if ((lane & 15) == 0) {     // lane 0 -> rows 0..7, lane 16 -> rows 8..15
        const float inv = inv_norm[0];
#pragma unroll
        for (int v = 0; v < 8; ++v)
            score[tile * 16 + hi * 8 + v] = tanhf(acc[v] * inv);
    }
}

// ---------------- 3) new_id init ----------------
__global__ void k_fill_newid(int* __restrict__ new_id) {
    int i = blockIdx.x * blockDim.x + threadIdx.x;
    if (i < N_TOT) new_id[i] = -1;
}

// ---------------- 4) per-graph full bitonic sort in LDS (256KB of the 320KB/WGP) ----------------
// Scores are DMA'd straight into the HIGH dword of each 64-bit key slot
// (little-endian: byte +4), then transformed in place:
// key = (~flip(score) << 32) | index  => ascending sort == descending score, stable by index.
__global__ void k_sort_seg(const float* __restrict__ score,
                           int* __restrict__ perm_ws,
                           float* __restrict__ score_perm_out,
                           int* __restrict__ perm_out,
                           int* __restrict__ batch_out,
                           int* __restrict__ new_id) {
    extern __shared__ unsigned long long keys[];
    const int seg = blockIdx.x;
    const int t = threadIdx.x, nt = blockDim.x;

    // phase 1: async global->LDS DMA of raw score bits; pads written directly
    for (int i = t; i < SORT_P2; i += nt) {
        if (i < N_PER) {
            async_load_b32_to_lds(lds_offset_of(&keys[i]) + 4u,
                                  score + seg * N_PER + i);
        } else {
            keys[i] = ~0ULL;            // pads sort to the back
        }
    }
    wait_asynccnt0();
    __syncthreads();

    // phase 2: in-place transform raw bits -> 64-bit sort key
    for (int i = t; i < N_PER; i += nt) {
        unsigned u = (unsigned)(keys[i] >> 32);
        unsigned asc = (u >> 31) ? ~u : (u | 0x80000000u);
        keys[i] = ((unsigned long long)(~asc) << 32) | (unsigned)i;
    }
    __syncthreads();

    for (int k = 2; k <= SORT_P2; k <<= 1) {
        for (int j = k >> 1; j > 0; j >>= 1) {
            for (int i = t; i < SORT_P2; i += nt) {
                int ixj = i ^ j;
                if (ixj > i) {
                    unsigned long long a = keys[i], b = keys[ixj];
                    bool up = ((i & k) == 0);
                    if ((a > b) == up) { keys[i] = b; keys[ixj] = a; }
                }
            }
            __syncthreads();
        }
    }

    for (int i = t; i < K_KEEP; i += nt) {
        int local = (int)(unsigned)(keys[i] & 0xFFFFFFFFu);
        int g = seg * N_PER + local;
        int j = seg * K_KEEP + i;
        perm_ws[j] = g;
        perm_out[j] = g;
        batch_out[j] = seg;
        score_perm_out[j] = score[g];
        new_id[g] = j;
    }
}

// ---------------- 5) x_out = x[perm] * score_perm ----------------
__global__ void k_gather_x(const float* __restrict__ x, const int* __restrict__ perm,
                           const float* __restrict__ score_perm, float* __restrict__ x_out) {
    int tid = blockIdx.x * blockDim.x + threadIdx.x;
    int row = tid >> 6, q = tid & 63;           // 64 x float4 per 256-wide row
    if (row >= BK) return;
    int src = perm[row];
    float s = score_perm[row];
    const f4* sp = (const f4*)(x + (size_t)src * C_DIM) + q;
    f4 v = __builtin_nontemporal_load(sp);      // read-once: TH=NT
    v *= s;
    __builtin_nontemporal_store(v, (f4*)(x_out + (size_t)row * C_DIM) + q);
}

// ---------------- 6) edge valid counts per block ----------------
__global__ void k_edge_count(const int* __restrict__ ei, const int* __restrict__ new_id,
                             int* __restrict__ block_counts) {
    __shared__ int tmp[1024];
    int i = blockIdx.x * blockDim.x + threadIdx.x;
    int v = 0;
    if (i < N_TOT) {
        int r = new_id[ei[i]];
        int c = new_id[ei[N_TOT + i]];
        v = (r >= 0 && c >= 0) ? 1 : 0;
    }
    tmp[threadIdx.x] = v;
    __syncthreads();
    for (int off = blockDim.x >> 1; off > 0; off >>= 1) {
        if ((int)threadIdx.x < off) tmp[threadIdx.x] += tmp[threadIdx.x + off];
        __syncthreads();
    }
    if (threadIdx.x == 0) block_counts[blockIdx.x] = tmp[0];
}

// ---------------- 7) scan block counts (tiny) ----------------
__global__ void k_scan_blocks(const int* __restrict__ counts, int* __restrict__ offs,
                              int* __restrict__ nval_ws, int* __restrict__ out_num_valid,
                              int nblocks) {
    if (threadIdx.x == 0 && blockIdx.x == 0) {
        int acc = 0;
        for (int i = 0; i < nblocks; ++i) { offs[i] = acc; acc += counts[i]; }
        nval_ws[0] = acc;
        out_num_valid[0] = acc;
    }
}

// ---------------- 8) stable scatter of edge_index + src map ----------------
__global__ void k_edge_scatter(const int* __restrict__ ei, const int* __restrict__ new_id,
                               const int* __restrict__ offs, const int* __restrict__ nval_ws,
                               int* __restrict__ eio, int* __restrict__ edge_src) {
    __shared__ int tmp[1024];
    int i = blockIdx.x * blockDim.x + threadIdx.x;
    int r = -1, c = -1, v = 0;
    if (i < N_TOT) {
        r = new_id[ei[i]];
        c = new_id[ei[N_TOT + i]];
        v = (r >= 0 && c >= 0) ? 1 : 0;
    }
    tmp[threadIdx.x] = v;
    __syncthreads();
    for (int off = 1; off < (int)blockDim.x; off <<= 1) {   // Hillis-Steele inclusive scan
        int s = ((int)threadIdx.x >= off) ? tmp[threadIdx.x - off] : 0;
        __syncthreads();
        tmp[threadIdx.x] += s;
        __syncthreads();
    }
    if (i < N_TOT) {
        int excl = tmp[threadIdx.x] - v;
        int gpv = offs[blockIdx.x] + excl;                  // global exclusive valid prefix
        int pos = v ? gpv : nval_ws[0] + (i - gpv);         // invalids stably after valids
        eio[pos] = v ? r : -1;
        eio[N_TOT + pos] = v ? c : -1;
        edge_src[pos] = v ? i : -1;
    }
}

// ---------------- 9) edge_attr_out streaming gather (410MB, NT loads+stores) ----------------
__global__ void k_edge_attr(const float* __restrict__ ea, const int* __restrict__ edge_src,
                            float* __restrict__ eao) {
    int tid = blockIdx.x * blockDim.x + threadIdx.x;
    int p = tid >> 6, q = tid & 63;
    if (p >= N_TOT) return;
    int src = edge_src[p];
    f4 v = {0.0f, 0.0f, 0.0f, 0.0f};
    if (src >= 0) {
        const f4* sp = (const f4*)(ea + (size_t)src * C_DIM) + q;
        __builtin_prefetch(sp + 16, 0, 1);   // -> global_prefetch_b8
        v = __builtin_nontemporal_load(sp);
    }
    __builtin_nontemporal_store(v, (f4*)(eao + (size_t)p * C_DIM) + q);
}

// ---------------- launch ----------------
extern "C" void kernel_launch(void* const* d_in, const int* in_sizes, int n_in,
                              void* d_out, int out_size, void* d_ws, size_t ws_size,
                              hipStream_t stream) {
    (void)in_sizes; (void)n_in; (void)out_size; (void)ws_size;
    const float* x  = (const float*)d_in[0];
    const int*   ei = (const int*)d_in[1];
    const float* ea = (const float*)d_in[2];
    // d_in[3] = batch (unused: segments are fixed-size, batch_out = seg id)
    const float* w  = (const float*)d_in[4];

    float* out = (float*)d_out;
    // Tuple outputs concatenated flat in return order (ints bit-stored):
    const size_t X_OFF     = 0;                              // x_out        [BK*C]
    const size_t EI_OFF    = (size_t)BK * C_DIM;             // edge_index   [2*N]
    const size_t EA_OFF    = EI_OFF + 2 * (size_t)N_TOT;     // edge_attr    [N*C]
    const size_t BATCH_OFF = EA_OFF + (size_t)N_TOT * C_DIM; // batch_out    [BK]
    const size_t PERM_OFF  = BATCH_OFF + BK;                 // perm         [BK]
    const size_t SP_OFF    = PERM_OFF + BK;                  // score_perm   [BK]
    const size_t NV_OFF    = SP_OFF + BK;                    // num_valid    [1]

    char* wsp = (char*)d_ws;
    float* score   = (float*)wsp; wsp += sizeof(float) * N_TOT;
    float* inv_nrm = (float*)wsp; wsp += 256;
    int*   perm_ws = (int*)wsp;   wsp += sizeof(int) * BK;
    int*   new_id  = (int*)wsp;   wsp += sizeof(int) * N_TOT;
    int*   bcnt    = (int*)wsp;   wsp += sizeof(int) * 512;
    int*   boff    = (int*)wsp;   wsp += sizeof(int) * 512;
    int*   nval_ws = (int*)wsp;   wsp += 256;
    int*   esrc    = (int*)wsp;   wsp += sizeof(int) * N_TOT;

    k_wnorm<<<1, 256, 0, stream>>>(w, inv_nrm);
    k_score_wmma<<<N_TOT / 128, 256, 0, stream>>>(ea, w, inv_nrm, score);
    k_fill_newid<<<(N_TOT + 255) / 256, 256, 0, stream>>>(new_id);
    k_sort_seg<<<B_GRAPHS, 1024, SORT_P2 * sizeof(unsigned long long), stream>>>(
        score, perm_ws, out + SP_OFF, (int*)(out + PERM_OFF), (int*)(out + BATCH_OFF), new_id);
    k_gather_x<<<(BK * 64) / 256, 256, 0, stream>>>(x, perm_ws, out + SP_OFF, out + X_OFF);

    const int EB = 1024;
    const int nblocks = (N_TOT + EB - 1) / EB;
    k_edge_count<<<nblocks, EB, 0, stream>>>(ei, new_id, bcnt);
    k_scan_blocks<<<1, 32, 0, stream>>>(bcnt, boff, nval_ws, (int*)(out + NV_OFF), nblocks);
    k_edge_scatter<<<nblocks, EB, 0, stream>>>(ei, new_id, boff, nval_ws, (int*)(out + EI_OFF), esrc);
    k_edge_attr<<<((size_t)N_TOT * 64) / 256, 256, 0, stream>>>(ea, esrc, out + EA_OFF);
}
